// EntropicOTQuantileRegression_59184649339522
// MI455X (gfx1250) — matrix operand order repaired and checked by reading, not compiled, not run on hardware
//
#include <hip/hip_runtime.h>

typedef __attribute__((ext_vector_type(16))) _Float16 v16h;
typedef __attribute__((ext_vector_type(8)))  float    v8f;

#define N_PTS 1024
#define M_SMP 2048
#define DXF   48
#define DYF   8
#define HDIM  64
#define EPSV  1e-7f
#define INV_EPS 1e7f

#define LOG2E 1.4426950408889634f
#define LN2   0.6931471805599453f

// stable softplus using raw HW transcendentals (v_exp_f32 / v_log_f32 are base-2):
// softplus(x) = max(x,0) + ln2 * log2(1 + exp2(-|x| * log2e))
// exp2 arg <= 0 and log2 arg in [1,2] -> no range fixups needed.
__device__ __forceinline__ float softplusf(float x) {
    float t = __builtin_amdgcn_exp2f(fabsf(x) * (-LOG2E));
    float l = __builtin_amdgcn_logf(1.0f + t);
    return fmaf(l, LN2, fmaxf(x, 0.0f));
}

// exp(d / EPS) for d <= 0 via one mul + one v_exp_f32; exp2(-inf) = 0.
__device__ __forceinline__ float exp_inv_eps(float d) {
    return __builtin_amdgcn_exp2f(d * (INV_EPS * LOG2E));
}

// hxb[i,k] = b1[k] + sum_d X[i,d] * W1[d,k]       (W1 rows 0..47)
__global__ void hx_kernel(const float* __restrict__ X, const float* __restrict__ W1,
                          const float* __restrict__ b1, float* __restrict__ hxb) {
    int t = blockIdx.x * blockDim.x + threadIdx.x;
    if (t >= N_PTS * HDIM) return;
    int i = t >> 6, k = t & 63;
    float acc = b1[k];
    const float* xr = X + i * DXF;
    #pragma unroll
    for (int d = 0; d < DXF; ++d) acc += xr[d] * W1[d * HDIM + k];
    hxb[t] = acc;
}

// hu[j,k] = sum_d U[j,d] * W1[48+d,k]             (W1 rows 48..55)
__global__ void hu_kernel(const float* __restrict__ U, const float* __restrict__ W1,
                          float* __restrict__ hu) {
    int t = blockIdx.x * blockDim.x + threadIdx.x;
    if (t >= M_SMP * HDIM) return;
    int j = t >> 6, k = t & 63;
    float acc = 0.f;
    const float* ur = U + j * DYF;
    #pragma unroll
    for (int d = 0; d < DYF; ++d) acc += ur[d] * W1[(DXF + d) * HDIM + k];
    hu[t] = acc;
}

// One wave (32 lanes) per data point i. Streams j in chunks of 16 using
// v_wmma_f32_16x16x32_f16 for h1 @ W2 (the 17 GFLOP hot loop).
__global__ __launch_bounds__(128) void psi_kernel(
    const float* __restrict__ hxb, const float* __restrict__ hu,
    const float* __restrict__ U,   const float* __restrict__ Y,
    const float* __restrict__ W2,  const float* __restrict__ b2,
    const float* __restrict__ W3,  const float* __restrict__ b3,
    float* __restrict__ psi)
{
    const int lane = threadIdx.x & 31;
    const int wave = threadIdx.x >> 5;
    const int i    = blockIdx.x * 4 + wave;
    const int n    = lane & 15;   // column-within-tile / A row
    const int g    = lane >> 4;   // half-wave group

    // ---- W2 as resident fp16 B tiles (ISA 16-bit B layout: K = ks*32 + g*8 + (e<8 ? e : e+8)) ----
    v16h Bt[2][4];
    #pragma unroll
    for (int ks = 0; ks < 2; ++ks)
        #pragma unroll
        for (int t = 0; t < 4; ++t)
            #pragma unroll
            for (int e = 0; e < 16; ++e) {
                int k = ks * 32 + g * 8 + (e < 8 ? e : e + 8);
                Bt[ks][t][e] = (_Float16)W2[k * HDIM + t * 16 + n];
            }

    float b2c[4], w3c[4];
    #pragma unroll
    for (int t = 0; t < 4; ++t) { b2c[t] = b2[t * 16 + n]; w3c[t] = W3[t * 16 + n]; }
    const float b3v = b3[0];

    // ---- hx[i,:] (+b1 folded) pre-permuted into A-layout order ----
    // A layout (16-bit 16x32): lane group g, element e -> k = g*8 + ks*32 + (e<8 ? e : e+8)
    float hxv[2][16];
    const float* hxp = hxb + i * HDIM;
    #pragma unroll
    for (int ks = 0; ks < 2; ++ks) {
        int k0 = g * 8 + ks * 32;
        #pragma unroll
        for (int e = 0; e < 16; ++e)
            hxv[ks][e] = hxp[k0 + (e < 8 ? e : e + 8)];
    }

    float yv[8];
    #pragma unroll
    for (int d = 0; d < 8; ++d) yv[d] = Y[i * DYF + d];

    float mx = -INFINITY, s = 0.f;   // streaming logsumexp state (scaled by 1/EPS)

    for (int c = 0; c < M_SMP / 16; ++c) {
        const int j0 = c * 16;

        v8f acc[4];
        #pragma unroll
        for (int t = 0; t < 4; ++t) acc[t] = (v8f){0.f,0.f,0.f,0.f,0.f,0.f,0.f,0.f};

        // ---- build h1 fp16 A tiles on the fly, 2 K-steps x 4 N-tiles of WMMA ----
        #pragma unroll
        for (int ks = 0; ks < 2; ++ks) {
            const float* hup = hu + (j0 + n) * HDIM + g * 8 + ks * 32;
            v16h A;
            #pragma unroll
            for (int e = 0; e < 8; ++e)
                A[e] = (_Float16)softplusf(hxv[ks][e] + hup[e]);
            #pragma unroll
            for (int e = 8; e < 16; ++e)
                A[e] = (_Float16)softplusf(hxv[ks][e] + hup[e + 8]);

            #pragma unroll
            for (int t = 0; t < 4; ++t)
                acc[t] = __builtin_amdgcn_wmma_f32_16x16x32_f16(
                    false, A, false, Bt[ks][t], (short)0, acc[t], false, false);
        }

        // ---- phi partials: softplus(h2_pre + b2) * W3, per-lane over 4 col tiles ----
        float part[8];
        #pragma unroll
        for (int r = 0; r < 8; ++r) {
            float p = 0.f;
            #pragma unroll
            for (int t = 0; t < 4; ++t)
                p += softplusf(acc[t][r] + b2c[t]) * w3c[t];
            part[r] = p;
        }
        // sum across the 16 lanes of each half-wave (columns of the D tile)
        #pragma unroll
        for (int r = 0; r < 8; ++r) {
            float v = part[r];
            v += __shfl_xor(v, 1, 32);
            v += __shfl_xor(v, 2, 32);
            v += __shfl_xor(v, 4, 32);
            v += __shfl_xor(v, 8, 32);
            part[r] = v;
        }

        // ---- streaming LSE update: lanes (n<8) each own one j of this chunk ----
        if (n < 8) {
            const int r = n;                 // D-tile row; j = j0 + r + 8*g
            float phi = b3v;
            #pragma unroll
            for (int q = 0; q < 8; ++q) phi += (r == q) ? part[q] : 0.f;

            const int jj = j0 + r + 8 * g;
            const float* ur = U + jj * DYF;
            float cost = 0.f;
            #pragma unroll
            for (int d = 0; d < 8; ++d) cost += yv[d] * ur[d];

            const float slack = cost - phi;
            const float m2 = fmaxf(mx, slack);
            const float ea = (mx    >= m2) ? 1.f : exp_inv_eps(mx    - m2);
            const float eb = (slack >= m2) ? 1.f : exp_inv_eps(slack - m2);
            s = s * ea + eb;
            mx = m2;
        }
    }

    // ---- combine (mx, s) across all 32 lanes; -inf lanes contribute nothing ----
    #pragma unroll
    for (int off = 1; off < 32; off <<= 1) {
        const float mo = __shfl_xor(mx, off, 32);
        const float so = __shfl_xor(s,  off, 32);
        const float m2 = fmaxf(mx, mo);
        const float ea = (mx >= m2) ? 1.f : exp_inv_eps(mx - m2);
        const float eb = (mo >= m2) ? 1.f : exp_inv_eps(mo - m2);
        s  = s * ea + so * eb;
        mx = m2;
    }

    if (lane == 0) {
        float ln_s = __builtin_amdgcn_logf(s) * LN2;           // ln(s)
        psi[i] = EPSV * (ln_s - 7.6246189861593985f) + mx;     // ln(2048) = 7.62461898...
    }
}

extern "C" void kernel_launch(void* const* d_in, const int* in_sizes, int n_in,
                              void* d_out, int out_size, void* d_ws, size_t ws_size,
                              hipStream_t stream) {
    const float* X  = (const float*)d_in[0];
    const float* U  = (const float*)d_in[1];
    const float* Y  = (const float*)d_in[2];
    const float* W1 = (const float*)d_in[3];
    const float* b1 = (const float*)d_in[4];
    const float* W2 = (const float*)d_in[5];
    const float* b2 = (const float*)d_in[6];
    const float* W3 = (const float*)d_in[7];
    const float* b3 = (const float*)d_in[8];
    float* psi = (float*)d_out;

    float* hxb = (float*)d_ws;              // [N, 64]
    float* hu  = hxb + N_PTS * HDIM;        // [M, 64]   total 768 KB

    hipLaunchKernelGGL(hx_kernel, dim3((N_PTS * HDIM + 255) / 256), dim3(256), 0, stream,
                       X, W1, b1, hxb);
    hipLaunchKernelGGL(hu_kernel, dim3((M_SMP * HDIM + 255) / 256), dim3(256), 0, stream,
                       U, W1, hu);
    hipLaunchKernelGGL(psi_kernel, dim3(N_PTS / 4), dim3(128), 0, stream,
                       hxb, hu, U, Y, W2, b2, W3, b3, psi);
}
